// ActivateLSDMModule_11158325035643
// MI455X (gfx1250) — matrix operand order repaired and checked by reading, not compiled, not run on hardware
//
#include <hip/hip_runtime.h>
#include <hip/hip_bf16.h>

// Problem dims (fixed by reference)
#define B_   64
#define S_   1024
#define I_   256
#define H_   512
#define G4_  2048            // 4*H
#define BS_  (B_ * S_)

typedef __attribute__((ext_vector_type(16))) _Float16 v16h;
typedef __attribute__((ext_vector_type(8)))  _Float16 v8h;
typedef __attribute__((ext_vector_type(8)))  float    v8f;

static __device__ __forceinline__ float sigm_(float x) {
    return 1.0f / (1.0f + __expf(-x));
}
static __device__ __forceinline__ float tanh_(float x) {
    float t = __expf(-2.0f * fabsf(x));
    float r = (1.0f - t) / (1.0f + t);
    return x < 0.0f ? -r : r;
}

// ---- WMMA fragment helpers (v_wmma_f32_16x16x32_f16) -----------------------
// A: 16(M)x32(K) f16, row-major source. Lane l<16 holds row M=l, K {kb..kb+15}
// with kb = 0; lanes 16-31 hold same rows with kb = 16. Two 16B loads per lane.
static __device__ __forceinline__ v16h ldA(const _Float16* a, size_t lda, int lane) {
    int row = lane & 15;
    int kb  = (lane >> 4) << 4;            // 0 or 16
    const _Float16* p = a + (size_t)row * lda + kb;
    v8h lo = *(const v8h*)p;               // K = kb .. kb+7
    v8h hi = *(const v8h*)(p + 8);         // K = kb+8 .. kb+15
    v16h v;
#pragma unroll
    for (int i = 0; i < 8; ++i) { v[i] = lo[i]; v[i + 8] = hi[i]; }
    return v;
}
// B: 32(K)x16(N) f16 sourced from a transposed (N x K row-major) matrix.
// Lane l<16 holds col N=l, K=0..15 contiguous (32B); lanes 16-31 hold K=16..31.
static __device__ __forceinline__ v16h ldB(const _Float16* bt, size_t ldb, int lane) {
    int col = lane & 15;
    int kb  = (lane >> 4) << 4;            // 0 or 16
    return *(const v16h*)(bt + (size_t)col * ldb + kb);
}
static __device__ __forceinline__ v8f mma(v16h a, v16h b, v8f c) {
    return __builtin_amdgcn_wmma_f32_16x16x32_f16(false, a, false, b, (short)0, c,
                                                  false, false);
}
static __device__ __forceinline__ v8f gemmK(const _Float16* A, size_t lda,
                                            const _Float16* Bt, size_t ldb,
                                            int K, int lane) {
    v8f acc = {};
    for (int k = 0; k < K; k += 32)
        acc = mma(ldA(A + k, lda, lane), ldB(Bt + k, ldb, lane), acc);
    return acc;
}

// ---- Prep kernels ----------------------------------------------------------
__global__ void k_init(const float* __restrict__ h0, _Float16* __restrict__ hbuf,
                       unsigned* __restrict__ cnt) {
    int idx = blockIdx.x * blockDim.x + threadIdx.x;
    if (idx < B_ * H_) hbuf[idx] = (_Float16)h0[idx];
    if (idx == 0) *cnt = 0u;
}

__global__ void k_cvtx(const float* __restrict__ x, _Float16* __restrict__ Xh,
                       float* __restrict__ z32, _Float16* __restrict__ z16) {
    int idx = blockIdx.x * blockDim.x + threadIdx.x;
    if (idx >= BS_ * I_) return;
    float v = x[idx];
    Xh[idx]  = (_Float16)v;
    z32[idx] = v;
    z16[idx] = (_Float16)v;
}

// Transpose+convert f32 (R x C) -> f16 (C x R)
__global__ void k_packT(const float* __restrict__ in, _Float16* __restrict__ out,
                        int R, int C) {
    int idx = blockIdx.x * blockDim.x + threadIdx.x;
    if (idx >= R * C) return;
    int r = idx / C, c = idx % C;
    out[(size_t)c * R + r] = (_Float16)in[idx];
}

// Gate weights: W_g is (H+I=768, H=512) f32; rows 0..511 multiply h, 512..767 multiply x.
// Build WhT (2048 x 512) and WxT (2048 x 256), both N-major f16 (row n = gate*512+col).
__global__ void k_packg(const float* __restrict__ Wf, const float* __restrict__ Wi,
                        const float* __restrict__ Wc, const float* __restrict__ Wo,
                        _Float16* __restrict__ WhT, _Float16* __restrict__ WxT) {
    int idx = blockIdx.x * blockDim.x + threadIdx.x;
    if (idx >= G4_ * 768) return;
    int n = idx / 768, kk = idx % 768;
    int g = n >> 9, col = n & 511;
    const float* W = (g == 0) ? Wf : (g == 1) ? Wi : (g == 2) ? Wc : Wo;
    if (kk < 512) WhT[(size_t)n * H_ + kk]        = (_Float16)W[(size_t)kk * H_ + col];
    else          WxT[(size_t)n * I_ + (kk - 512)] = (_Float16)W[(size_t)kk * H_ + col];
}

// ---- NeuralODE phase (parallel over 65536 rows) ----------------------------
// Hid = tanh(z16 @ A + ba)   : M=BS, N=512, K=256
__global__ void __launch_bounds__(256) k_hid(const _Float16* __restrict__ z16,
                                             const _Float16* __restrict__ oAT,
                                             const float* __restrict__ ba,
                                             _Float16* __restrict__ Hid) {
    int wid  = (blockIdx.x * blockDim.x + threadIdx.x) >> 5;
    int lane = threadIdx.x & 31;
    int mt = wid >> 5, nt = wid & 31;              // 4096 x 32 tiles
    int m0 = mt << 4, n0 = nt << 4;
    v8f acc = gemmK(z16 + (size_t)m0 * I_, I_, oAT + (size_t)n0 * I_, I_, I_, lane);
    int n  = n0 + (lane & 15);
    int rb = (lane >> 4) << 3;
    float bav = ba[n];
#pragma unroll
    for (int r = 0; r < 8; ++r) {
        int m = m0 + rb + r;
        Hid[(size_t)m * H_ + n] = (_Float16)tanh_(acc[r] + bav);
    }
}

// z += dt * (Hid @ Bm)       : M=BS, N=256, K=512
__global__ void __launch_bounds__(256) k_zupd(const _Float16* __restrict__ Hid,
                                              const _Float16* __restrict__ oBT,
                                              float* __restrict__ z32,
                                              _Float16* __restrict__ z16) {
    int wid  = (blockIdx.x * blockDim.x + threadIdx.x) >> 5;
    int lane = threadIdx.x & 31;
    int mt = wid >> 4, nt = wid & 15;              // 4096 x 16 tiles
    int m0 = mt << 4, n0 = nt << 4;
    v8f acc = gemmK(Hid + (size_t)m0 * H_, H_, oBT + (size_t)n0 * H_, H_, H_, lane);
    int n  = n0 + (lane & 15);
    int rb = (lane >> 4) << 3;
#pragma unroll
    for (int r = 0; r < 8; ++r) {
        size_t idx = (size_t)(m0 + rb + r) * I_ + n;
        float z = z32[idx] + 0.25f * acc[r];       // dt = 1/ODE_STEPS
        z32[idx] = z;
        z16[idx] = (_Float16)z;
    }
}

// Pw = P @ wode              : M=BS, N=512, K=256 (f32 output)
__global__ void __launch_bounds__(256) k_pw(const _Float16* __restrict__ P,
                                            const _Float16* __restrict__ wOT,
                                            float* __restrict__ Pw) {
    int wid  = (blockIdx.x * blockDim.x + threadIdx.x) >> 5;
    int lane = threadIdx.x & 31;
    int mt = wid >> 5, nt = wid & 31;
    int m0 = mt << 4, n0 = nt << 4;
    v8f acc = gemmK(P + (size_t)m0 * I_, I_, wOT + (size_t)n0 * I_, I_, I_, lane);
    int n  = n0 + (lane & 15);
    int rb = (lane >> 4) << 3;
#pragma unroll
    for (int r = 0; r < 8; ++r)
        Pw[(size_t)(m0 + rb + r) * H_ + n] = acc[r];
}

// ---- Persistent sequential scan -------------------------------------------
// 32 blocks x 128 threads (4 waves). Block owns hidden cols [16*blockIdx, +16);
// wave owns batch rows [16*waveId, +16). Each wave computes the f/i/c/o tiles
// for its (b,n) slice, so the LSTM pointwise update is wave-local and C stays
// in registers for the entire scan. h is double-buffered in global f16 with a
// device-scope barrier between steps. Gate weights live in LDS (96 KB/block).
__global__ void __launch_bounds__(128) k_seq(
        const _Float16* __restrict__ Xh, const _Float16* __restrict__ WhT,
        const _Float16* __restrict__ WxT,
        const float* __restrict__ bf, const float* __restrict__ bi,
        const float* __restrict__ bc, const float* __restrict__ bo,
        const float* __restrict__ Pw, const float* __restrict__ C0,
        _Float16* __restrict__ hbuf, float* __restrict__ out,
        unsigned* __restrict__ cnt) {
    extern __shared__ _Float16 Bsh[];               // [4 gates][16 cols][768 K]
    const int lane = threadIdx.x & 31;
    const int mt   = threadIdx.x >> 5;              // 0..3  (batch tile)
    const int n0   = blockIdx.x << 4;               // hidden col base
    const int m0   = mt << 4;

    // Stage this block's weight slice into LDS once (row n = gate*512 + n0+col).
    for (int idx = threadIdx.x; idx < 4 * 16 * 768; idx += blockDim.x) {
        int k = idx % 768, gc = idx / 768;
        int g = gc >> 4, col = gc & 15;
        _Float16 v;
        if (k < 512) v = WhT[(size_t)(g * H_ + n0 + col) * H_ + k];
        else         v = WxT[(size_t)(g * H_ + n0 + col) * I_ + (k - 512)];
        Bsh[idx] = v;
    }
    __syncthreads();

    const int nl = lane & 15, rb = (lane >> 4) << 3;
    const int n  = n0 + nl;
    const float bfv = bf[n], biv = bi[n], bcv = bc[n], bov = bo[n];
    float C[8];
#pragma unroll
    for (int r = 0; r < 8; ++r) C[r] = C0[(size_t)(m0 + rb + r) * H_ + n];

    const unsigned nb = gridDim.x;
    for (int s = 0; s < S_; ++s) {
        const _Float16* hp = hbuf + (size_t)(s & 1) * (B_ * H_);
        _Float16*       hn = hbuf + (size_t)((s & 1) ^ 1) * (B_ * H_);
        v8f af = {}, ai = {}, ac = {}, ao = {};

        const _Float16* Ah = hp + (size_t)m0 * H_;  // recurrent part, K = 512
#pragma unroll 4
        for (int k = 0; k < H_; k += 32) {
            v16h a = ldA(Ah + k, H_, lane);
            af = mma(a, ldB(Bsh + 0 * 16 * 768 + k, 768, lane), af);
            ai = mma(a, ldB(Bsh + 1 * 16 * 768 + k, 768, lane), ai);
            ac = mma(a, ldB(Bsh + 2 * 16 * 768 + k, 768, lane), ac);
            ao = mma(a, ldB(Bsh + 3 * 16 * 768 + k, 768, lane), ao);
        }
        const _Float16* Ax = Xh + ((size_t)m0 * S_ + s) * I_;  // input part, K = 256
#pragma unroll 4
        for (int k = 0; k < I_; k += 32) {
            v16h a = ldA(Ax + k, (size_t)S_ * I_, lane);
            af = mma(a, ldB(Bsh + 0 * 16 * 768 + 512 + k, 768, lane), af);
            ai = mma(a, ldB(Bsh + 1 * 16 * 768 + 512 + k, 768, lane), ai);
            ac = mma(a, ldB(Bsh + 2 * 16 * 768 + 512 + k, 768, lane), ac);
            ao = mma(a, ldB(Bsh + 3 * 16 * 768 + 512 + k, 768, lane), ao);
        }

#pragma unroll
        for (int r = 0; r < 8; ++r) {
            int b = m0 + rb + r;
            size_t row = (size_t)b * S_ + s;
            float fg = sigm_(af[r] + bfv);
            float ig = sigm_(ai[r] + biv);
            float cg = tanh_(ac[r] + bcv);
            float og = sigm_(ao[r] + bov + Pw[row * H_ + n]);
            float Cn = fg * C[r] + ig * cg;
            float hv = og * tanh_(Cn);
            C[r] = Cn;
            hn[(size_t)b * H_ + n] = (_Float16)hv;
            out[row * H_ + n] = hv;
            if (s == S_ - 1) {
                out[(size_t)B_ * S_ * H_ + (size_t)b * H_ + n] = hv;                 // hT
                out[(size_t)B_ * S_ * H_ + (size_t)B_ * H_ + (size_t)b * H_ + n] = Cn; // CT
            }
        }

        // Device-scope barrier: publish hn, wait for all 32 blocks.
        __threadfence();
        __syncthreads();
        if (threadIdx.x == 0) {
            __hip_atomic_fetch_add(cnt, 1u, __ATOMIC_ACQ_REL, __HIP_MEMORY_SCOPE_AGENT);
            unsigned tgt = (unsigned)(s + 1) * nb;
            while (__hip_atomic_load(cnt, __ATOMIC_ACQUIRE, __HIP_MEMORY_SCOPE_AGENT) < tgt)
                __builtin_amdgcn_s_sleep(1);
        }
        __syncthreads();
    }
}

// ---- Host launch -----------------------------------------------------------
extern "C" void kernel_launch(void* const* d_in, const int* in_sizes, int n_in,
                              void* d_out, int out_size, void* d_ws, size_t ws_size,
                              hipStream_t stream) {
    (void)in_sizes; (void)n_in; (void)out_size; (void)ws_size;
    const float* x    = (const float*)d_in[0];
    const float* h0   = (const float*)d_in[1];
    const float* C0   = (const float*)d_in[2];
    const float* Wf   = (const float*)d_in[3];
    const float* bf   = (const float*)d_in[4];
    const float* Wi   = (const float*)d_in[5];
    const float* bi   = (const float*)d_in[6];
    const float* Wc   = (const float*)d_in[7];
    const float* bc   = (const float*)d_in[8];
    const float* Wo   = (const float*)d_in[9];
    const float* bo   = (const float*)d_in[10];
    const float* wode = (const float*)d_in[11];
    const float* odeA = (const float*)d_in[12];
    const float* odba = (const float*)d_in[13];
    const float* odeB = (const float*)d_in[14];
    float* out = (float*)d_out;

    // Workspace carve-out (~196 MB). Pw (f32, BS x H) aliases the dead
    // z32+Hid region after the ODE phase finishes.
    char* ws = (char*)d_ws;
    size_t o = 0;
    auto take = [&](size_t bytes) -> char* {
        char* p = ws + o; o += (bytes + 255) & ~(size_t)255; return p;
    };
    _Float16* Xh  = (_Float16*)take((size_t)BS_ * I_ * 2);
    _Float16* z16 = (_Float16*)take((size_t)BS_ * I_ * 2);
    char*     pwz = take((size_t)BS_ * H_ * 4);          // 134 MB shared region
    float*    z32 = (float*)pwz;                         // BS x I f32 (first 64 MB)
    _Float16* Hid = (_Float16*)(pwz + (size_t)BS_ * I_ * 4); // BS x H f16 (next 64 MB)
    float*    Pw  = (float*)pwz;                         // BS x H f32 (aliases both)
    _Float16* WhT = (_Float16*)take((size_t)G4_ * H_ * 2);
    _Float16* WxT = (_Float16*)take((size_t)G4_ * I_ * 2);
    _Float16* oAT = (_Float16*)take((size_t)H_ * I_ * 2);
    _Float16* oBT = (_Float16*)take((size_t)I_ * H_ * 2);
    _Float16* wOT = (_Float16*)take((size_t)H_ * I_ * 2);
    _Float16* hbuf = (_Float16*)take((size_t)2 * B_ * H_ * 2);
    unsigned* cnt  = (unsigned*)take(256);

    // Prep
    k_init<<<(B_ * H_ + 255) / 256, 256, 0, stream>>>(h0, hbuf, cnt);
    k_cvtx<<<(BS_ * I_ + 255) / 256, 256, 0, stream>>>(x, Xh, z32, z16);
    k_packg<<<(G4_ * 768 + 255) / 256, 256, 0, stream>>>(Wf, Wi, Wc, Wo, WhT, WxT);
    k_packT<<<(I_ * 2 * I_ + 255) / 256, 256, 0, stream>>>(odeA, oAT, I_, 2 * I_);
    k_packT<<<(I_ * 2 * I_ + 255) / 256, 256, 0, stream>>>(odeB, oBT, 2 * I_, I_);
    k_packT<<<(I_ * H_ + 255) / 256, 256, 0, stream>>>(wode, wOT, I_, H_);

    // NeuralODE: 4 Euler steps (parallel over all B*S rows)
    for (int t = 0; t < 4; ++t) {
        k_hid<<<(BS_ / 16) * 32 * 32 / 256, 256, 0, stream>>>(z16, oAT, odba, Hid);
        k_zupd<<<(BS_ / 16) * 16 * 32 / 256, 256, 0, stream>>>(Hid, oBT, z32, z16);
    }
    k_pw<<<(BS_ / 16) * 32 * 32 / 256, 256, 0, stream>>>(z16, wOT, Pw);

    // Recurrent scan: 32 blocks x 128 threads, 96 KB dynamic LDS per block
    k_seq<<<32, 128, 4 * 16 * 768 * (int)sizeof(_Float16), stream>>>(
        Xh, WhT, WxT, bf, bi, bc, bo, Pw, C0, hbuf, out, cnt);
}
